// Attention_20418274525497
// MI455X (gfx1250) — matrix operand order repaired and checked by reading, not compiled, not run on hardware
//
#include <hip/hip_runtime.h>
#include <hip/hip_bf16.h>

// ---------------------------------------------------------------------------
// Problem constants (from reference)
// ---------------------------------------------------------------------------
#define WSZ       8
#define DIMC      256
#define DIM_HEAD  32
#define HEADSN    8
#define INNERC    256
#define SSN       64                 // tokens per window
#define D4N       64
#define HW        256                // H == W == 256
#define SCALE_F   0.17677669529663687f   // 32^-0.5
#define EPS_F     1e-5f

// CDNA5 WMMA vector types (wave32)
typedef __attribute__((ext_vector_type(16))) __bf16 v16bf;
typedef __attribute__((ext_vector_type(8)))  float  v8f;

union FragU { uint4 q[2]; v16bf v; };

__device__ __forceinline__ unsigned short f2bf(float f) {
    unsigned int u = __float_as_uint(f);
    u += 0x7FFFu + ((u >> 16) & 1u);           // round-to-nearest-even
    return (unsigned short)(u >> 16);
}

// ---------------------------------------------------------------------------
// Workspace layout (bytes)
//   [0,        16384)  : bias table fp32 [64][64]
//   [16384,   409600)  : w_qkv bf16, fragment-swizzled: 48 ntiles x 8 ksteps x 1KB
//   [409600,  540672)  : w_out bf16, fragment-swizzled: 16 ntiles x 8 ksteps x 1KB
// ---------------------------------------------------------------------------
#define WS_BIAS_OFF   0
#define WS_WQKV_OFF   16384
#define WS_WOUT_OFF   409600

// ---------------------------------------------------------------------------
// Kernel 0: dynamic position bias MLP -> bias[64][64] table (run once, tiny)
// ---------------------------------------------------------------------------
__device__ __forceinline__ void ln_relu64(float* h, const float* gg, const float* bb) {
    float m = 0.f;
    #pragma unroll 8
    for (int c = 0; c < 64; ++c) m += h[c];
    m *= (1.0f / 64.0f);
    float v = 0.f;
    #pragma unroll 8
    for (int c = 0; c < 64; ++c) { float d = h[c] - m; v += d * d; }
    v *= (1.0f / 64.0f);
    float rs = rsqrtf(v + EPS_F);
    #pragma unroll 8
    for (int c = 0; c < 64; ++c) {
        float t = (h[c] - m) * rs * gg[c] + bb[c];
        h[c] = t > 0.f ? t : 0.f;
    }
}

__global__ void dpb_kernel(const float* __restrict__ w1, const float* __restrict__ b1,
                           const float* __restrict__ g1, const float* __restrict__ e1,
                           const float* __restrict__ w2, const float* __restrict__ b2,
                           const float* __restrict__ g2, const float* __restrict__ e2,
                           const float* __restrict__ w3, const float* __restrict__ b3,
                           const float* __restrict__ g3, const float* __restrict__ e3,
                           const float* __restrict__ w4, const float* __restrict__ b4,
                           float* __restrict__ bias_out) {
    __shared__ float table[289];
    const int t = threadIdx.x;
    if (t < 289) {
        float ci = (float)(t / 17 - WSZ);
        float cj = (float)(t % 17 - WSZ);
        float h[64], h2[64];
        #pragma unroll 4
        for (int c = 0; c < 64; ++c) h[c] = ci * w1[c] + cj * w1[64 + c] + b1[c];
        ln_relu64(h, g1, e1);
        for (int c = 0; c < 64; ++c) {
            float a = b2[c];
            for (int d = 0; d < 64; ++d) a += h[d] * w2[d * 64 + c];
            h2[c] = a;
        }
        ln_relu64(h2, g2, e2);
        for (int c = 0; c < 64; ++c) {
            float a = b3[c];
            for (int d = 0; d < 64; ++d) a += h2[d] * w3[d * 64 + c];
            h[c] = a;
        }
        ln_relu64(h, g3, e3);
        float a = b4[0];
        for (int d = 0; d < 64; ++d) a += h[d] * w4[d];
        table[t] = a;
    }
    __syncthreads();
    for (int e = t; e < SSN * SSN; e += blockDim.x) {
        int i = e >> 6, j = e & 63;
        int r0 = (i >> 3) - (j >> 3) + (WSZ - 1);
        int r1 = (i & 7)  - (j & 7)  + (WSZ - 1);
        bias_out[e] = table[r0 * (2 * WSZ - 1) + r1];
    }
}

// ---------------------------------------------------------------------------
// Kernel 1: pre-swizzle weights into WMMA bf16 B-fragment layout.
// B-fragment (32x16-halves = 1KB): lane L, half i  ->  B[k][n] with
//   n = ntile*16 + (L&15),  k = kstep*32 + (L<16 ? i : 16+i)
// ---------------------------------------------------------------------------
__global__ void swizzle_kernel(const float* __restrict__ wqkv,
                               const float* __restrict__ wout,
                               unsigned short* __restrict__ wqkv_sw,
                               unsigned short* __restrict__ wout_sw) {
    const int NQ = 48 * 8 * 512;   // 196608 halves
    const int NO = 16 * 8 * 512;   //  65536 halves
    int gidx = blockIdx.x * 256 + threadIdx.x;
    if (gidx < NQ) {
        int frag = gidx >> 9, rem = gidx & 511;
        int L = rem >> 4, i = rem & 15;
        int nt = frag >> 3, ks = frag & 7;
        int N = nt * 16 + (L & 15);
        int K = ks * 32 + ((L < 16) ? i : (16 + i));
        wqkv_sw[gidx] = f2bf(wqkv[N * DIMC + K]);        // B[k][n] = w_qkv[n][k]
    } else if (gidx < NQ + NO) {
        int gg = gidx - NQ;
        int frag = gg >> 9, rem = gg & 511;
        int L = rem >> 4, i = rem & 15;
        int nt = frag >> 3, ks = frag & 7;
        int N = nt * 16 + (L & 15);
        int K = ks * 32 + ((L < 16) ? i : (16 + i));
        wout_sw[gg] = f2bf(wout[N * DIMC + K]);
    }
}

// ---------------------------------------------------------------------------
// LDS layout (dynamic shared, ~226 KB; CDNA5 WGP allows up to 320 KB)
// ---------------------------------------------------------------------------
#define SM_Q_OFF      0            // bf16 [64][264]  (aliases Xf fp32 [64][260] in phase A)
#define SM_K_OFF      33792        // bf16 [64][264]
#define SM_VT_OFF     67584        // bf16 [256][72]  (V transposed: channel-major)
#define SM_XN_OFF     104448       // bf16 [64][264]  (LN'd input; aliased by O in phase C)
#define SM_P_OFF      138240       // bf16 [8 heads][64][72]
#define SM_BIAS_OFF   211968       // fp32 [64][64]
#define SM_PART_OFF   228352       // fp32 [256][2]
#define SM_STAT_OFF   230400       // fp32 [64][2]
#define SMEM_BYTES    230912

#define XSTR 264   // halves
#define PSTR 72    // halves
#define VSTR 72    // halves
#define XFSTR 260  // floats

__device__ __forceinline__ void loadA(FragU& A, const unsigned short* base, int stride,
                                      int mbase, int colbase, int laneMod, int half) {
    // 16-bit A layout (contiguous, matches B): lanes 0-15: K {kb..kb+15}, kb=colbase
    //                               lanes 16-31: kb=colbase+16
    int row = mbase + laneMod;
    int kb  = colbase + half * 16;
    const unsigned short* p = base + row * stride + kb;
    A.q[0] = *(const uint4*)(p);
    A.q[1] = *(const uint4*)(p + 8);
}

__device__ __forceinline__ void loadB_lds(FragU& B, const unsigned short* base, int stride,
                                          int nbase, int kcolbase, int laneMod, int half) {
    int n    = nbase + laneMod;
    int koff = kcolbase + half * 16;
    const unsigned short* p = base + n * stride + koff;
    B.q[0] = *(const uint4*)(p);
    B.q[1] = *(const uint4*)(p + 8);
}

__device__ __forceinline__ void loadB_gbl(FragU& B, const unsigned short* frag, int lane) {
    const unsigned short* p = frag + lane * 16;
    B.q[0] = *(const uint4*)(p);
    B.q[1] = *(const uint4*)(p + 8);
}

// ---------------------------------------------------------------------------
// Kernel 2: fully fused windowed attention. 1 workgroup = 1 window (8 wave32s)
// ---------------------------------------------------------------------------
__global__ __launch_bounds__(256, 1)
void attn_win_kernel(const float* __restrict__ x,
                     const float* __restrict__ gln, const float* __restrict__ bln,
                     const float* __restrict__ b_out,
                     const float* __restrict__ bias_ws,
                     const unsigned short* __restrict__ wqkv_sw,
                     const unsigned short* __restrict__ wout_sw,
                     float* __restrict__ out) {
    extern __shared__ char smem[];
    float*          Xf    = (float*)(smem + SM_Q_OFF);
    unsigned short* Qs    = (unsigned short*)(smem + SM_Q_OFF);
    unsigned short* Ks    = (unsigned short*)(smem + SM_K_OFF);
    unsigned short* Vt    = (unsigned short*)(smem + SM_VT_OFF);
    unsigned short* XN    = (unsigned short*)(smem + SM_XN_OFF);  // also O
    unsigned short* Pall  = (unsigned short*)(smem + SM_P_OFF);
    float*          biasS = (float*)(smem + SM_BIAS_OFF);
    float*          part  = (float*)(smem + SM_PART_OFF);
    float*          stat  = (float*)(smem + SM_STAT_OFF);

    const int tid  = threadIdx.x;
    // wave id is uniform per wave -> force into SGPR so tile decode / epilogue
    // selection compiles to scalar branches instead of EXEC-masked paths.
    const int wave = __builtin_amdgcn_readfirstlane(tid >> 5);
    const int lane = tid & 31;
    const int laneMod = lane & 15;
    const int half = lane >> 4;

    const int blk = blockIdx.x;
    const int bb  = blk >> 10;                 // batch
    const int r   = blk & 1023;
    const int y0  = (r >> 5) * WSZ;
    const int x0  = (r & 31) * WSZ;

    // ---------------- Phase A: load window + channel LayerNorm -> XN (bf16) --
    {
        const int pix = tid & 63, dsub = tid >> 6;
        const int py = pix >> 3, px = pix & 7;
        const size_t gbase = ((size_t)bb * DIMC) * (HW * HW) + (size_t)(y0 + py) * HW + (x0 + px);
        #pragma unroll 4
        for (int it = 0; it < 64; ++it) {
            int d = it * 4 + dsub;
            Xf[pix * XFSTR + d] = x[gbase + (size_t)d * (HW * HW)];
        }
        #pragma unroll
        for (int k = 0; k < 16; ++k) biasS[tid * 16 + k] = bias_ws[tid * 16 + k];
    }
    __syncthreads();
    {
        const int pix = tid >> 2, qq = tid & 3;
        float s = 0.f, s2 = 0.f;
        #pragma unroll 8
        for (int dd = 0; dd < 64; ++dd) {
            float v = Xf[pix * XFSTR + qq * 64 + dd];
            s += v; s2 += v * v;
        }
        part[tid * 2] = s; part[tid * 2 + 1] = s2;
    }
    __syncthreads();
    if (tid < 64) {
        float s = 0.f, s2 = 0.f;
        #pragma unroll
        for (int qq = 0; qq < 4; ++qq) {
            s  += part[(tid * 4 + qq) * 2];
            s2 += part[(tid * 4 + qq) * 2 + 1];
        }
        float m = s * (1.0f / DIMC);
        float v = s2 * (1.0f / DIMC) - m * m;
        stat[tid * 2] = m;
        stat[tid * 2 + 1] = rsqrtf(v + EPS_F);
    }
    __syncthreads();
    {
        const int pix = tid >> 2, qq = tid & 3;
        const float m = stat[pix * 2], rs = stat[pix * 2 + 1];
        #pragma unroll 8
        for (int dd = 0; dd < 64; ++dd) {
            int d = qq * 64 + dd;
            float v = (Xf[pix * XFSTR + d] - m) * rs * gln[d] + bln[d];
            XN[pix * XSTR + d] = f2bf(v);
        }
    }
    __syncthreads();

    // ---------------- Phase B: QKV GEMM (64x768 = XN @ Wqkv^T) via WMMA -----
    // 192 output tiles (4 M x 48 N) over 8 waves; K=256 in 8 steps.
    // B fragments are software-pipelined one step ahead of the wmma chain.
    for (int t = wave; t < 192; t += 8) {
        const int mt = t & 3, nt = t >> 2;
        v8f acc = {};
        const unsigned short* wb = wqkv_sw + (size_t)(nt * 8) * 512;
        FragU Bc, Bn;
        loadB_gbl(Bc, wb, lane);
        #pragma unroll
        for (int ks = 0; ks < 8; ++ks) {
            FragU A;
            loadA(A, XN, XSTR, mt * 16, ks * 32, laneMod, half);
            if (ks < 7) {
                loadB_gbl(Bn, wb + (ks + 1) * 512, lane);
                if (ks < 6) __builtin_prefetch(wb + (ks + 2) * 512 + lane * 16, 0, 3);
            }
            acc = __builtin_amdgcn_wmma_f32_16x16x32_bf16(false, A.v, false, Bc.v,
                                                          (short)0, acc, false, false);
            Bc = Bn;
        }
        const int Ncol = nt * 16;
        const int sec = Ncol >> 8;            // 0=Q 1=K 2=V (scalar branch: nt is SGPR)
        const int col = (Ncol & 255) + laneMod;
        if (sec == 0) {
            #pragma unroll
            for (int i = 0; i < 8; ++i)
                Qs[(mt * 16 + i + half * 8) * XSTR + col] = f2bf(acc[i] * SCALE_F);
        } else if (sec == 1) {
            #pragma unroll
            for (int i = 0; i < 8; ++i)
                Ks[(mt * 16 + i + half * 8) * XSTR + col] = f2bf(acc[i]);
        } else {
            // Vt row (channel-major): 8 consecutive halves -> single ds_store_b128
            unsigned int p0 = (unsigned int)f2bf(acc[0]) | ((unsigned int)f2bf(acc[1]) << 16);
            unsigned int p1 = (unsigned int)f2bf(acc[2]) | ((unsigned int)f2bf(acc[3]) << 16);
            unsigned int p2 = (unsigned int)f2bf(acc[4]) | ((unsigned int)f2bf(acc[5]) << 16);
            unsigned int p3 = (unsigned int)f2bf(acc[6]) | ((unsigned int)f2bf(acc[7]) << 16);
            *(uint4*)(Vt + col * VSTR + mt * 16 + half * 8) = make_uint4(p0, p1, p2, p3);
        }
    }
    __syncthreads();

    // ---------------- Phase C: attention, one wave per head -----------------
    {
        const int h = wave;                          // SGPR
        const int cb = h * DIM_HEAD;
        unsigned short* P = Pall + h * (SSN * PSTR);

        for (int mt = 0; mt < 4; ++mt) {
            FragU aQ;
            loadA(aQ, Qs, XSTR, mt * 16, cb, laneMod, half);
            v8f s[4];
            #pragma unroll
            for (int nt = 0; nt < 4; ++nt) {
                FragU bK;
                loadB_lds(bK, Ks, XSTR, nt * 16, cb, laneMod, half);
                v8f z = {};
                s[nt] = __builtin_amdgcn_wmma_f32_16x16x32_bf16(false, aQ.v, false, bK.v,
                                                                (short)0, z, false, false);
                #pragma unroll
                for (int i = 0; i < 8; ++i)
                    s[nt][i] += biasS[(mt * 16 + i + half * 8) * 64 + nt * 16 + laneMod];
            }
            // register softmax; each row lives in one lane-half across 16 lanes x 4 tiles
            #pragma unroll
            for (int i = 0; i < 8; ++i) {
                float m = fmaxf(fmaxf(s[0][i], s[1][i]), fmaxf(s[2][i], s[3][i]));
                m = fmaxf(m, __shfl_xor(m, 1, 32));
                m = fmaxf(m, __shfl_xor(m, 2, 32));
                m = fmaxf(m, __shfl_xor(m, 4, 32));
                m = fmaxf(m, __shfl_xor(m, 8, 32));
                float e0 = __expf(s[0][i] - m), e1 = __expf(s[1][i] - m);
                float e2 = __expf(s[2][i] - m), e3 = __expf(s[3][i] - m);
                float sum = e0 + e1 + e2 + e3;
                sum += __shfl_xor(sum, 1, 32);
                sum += __shfl_xor(sum, 2, 32);
                sum += __shfl_xor(sum, 4, 32);
                sum += __shfl_xor(sum, 8, 32);
                float rs = 1.0f / sum;
                const int row = mt * 16 + i + half * 8;
                P[row * PSTR +  0 + laneMod] = f2bf(e0 * rs);
                P[row * PSTR + 16 + laneMod] = f2bf(e1 * rs);
                P[row * PSTR + 32 + laneMod] = f2bf(e2 * rs);
                P[row * PSTR + 48 + laneMod] = f2bf(e3 * rs);
            }
        }
        // O = P(64x64) @ V(64x32); write O into XN region, token-major
        for (int mt = 0; mt < 4; ++mt) {
            FragU aP0, aP1;
            loadA(aP0, P, PSTR, mt * 16, 0,  laneMod, half);
            loadA(aP1, P, PSTR, mt * 16, 32, laneMod, half);
            #pragma unroll
            for (int ct = 0; ct < 2; ++ct) {
                v8f acc = {};
                FragU bV0, bV1;
                loadB_lds(bV0, Vt, VSTR, cb + ct * 16, 0,  laneMod, half);
                loadB_lds(bV1, Vt, VSTR, cb + ct * 16, 32, laneMod, half);
                acc = __builtin_amdgcn_wmma_f32_16x16x32_bf16(false, aP0.v, false, bV0.v,
                                                              (short)0, acc, false, false);
                acc = __builtin_amdgcn_wmma_f32_16x16x32_bf16(false, aP1.v, false, bV1.v,
                                                              (short)0, acc, false, false);
                const int col = cb + ct * 16 + laneMod;
                #pragma unroll
                for (int i = 0; i < 8; ++i)
                    XN[(mt * 16 + i + half * 8) * XSTR + col] = f2bf(acc[i]);
            }
        }
    }
    __syncthreads();

    // ---------------- Phase D: output projection + windowed store -----------
    for (int t = wave; t < 64; t += 8) {
        const int mt = t & 3, nt = t >> 2;
        v8f acc = {};
        const unsigned short* wb = wout_sw + (size_t)(nt * 8) * 512;
        FragU Bc, Bn;
        loadB_gbl(Bc, wb, lane);
        #pragma unroll
        for (int ks = 0; ks < 8; ++ks) {
            FragU A;
            loadA(A, XN, XSTR, mt * 16, ks * 32, laneMod, half);
            if (ks < 7) {
                loadB_gbl(Bn, wb + (ks + 1) * 512, lane);
                if (ks < 6) __builtin_prefetch(wb + (ks + 2) * 512 + lane * 16, 0, 3);
            }
            acc = __builtin_amdgcn_wmma_f32_16x16x32_bf16(false, A.v, false, Bc.v,
                                                          (short)0, acc, false, false);
            Bc = Bn;
        }
        const int c = nt * 16 + laneMod;
        const float bo = b_out[c];
        const int tok0 = mt * 16 + half * 8;      // 8 consecutive tokens = one window row
        const int py = tok0 >> 3;
        float* op = out + (((size_t)bb * DIMC + c) * HW + (y0 + py)) * HW + x0;
        float4 lo = make_float4(acc[0] + bo, acc[1] + bo, acc[2] + bo, acc[3] + bo);
        float4 hi = make_float4(acc[4] + bo, acc[5] + bo, acc[6] + bo, acc[7] + bo);
        *(float4*)(op)     = lo;
        *(float4*)(op + 4) = hi;
    }
}

// ---------------------------------------------------------------------------
extern "C" void kernel_launch(void* const* d_in, const int* in_sizes, int n_in,
                              void* d_out, int out_size, void* d_ws, size_t ws_size,
                              hipStream_t stream) {
    (void)in_sizes; (void)n_in; (void)out_size; (void)ws_size;
    const float* x      = (const float*)d_in[0];
    const float* g      = (const float*)d_in[1];
    const float* b      = (const float*)d_in[2];
    const float* w_qkv  = (const float*)d_in[3];
    const float* w_out  = (const float*)d_in[4];
    const float* b_out  = (const float*)d_in[5];
    const float* dw1 = (const float*)d_in[6],  *db1 = (const float*)d_in[7];
    const float* dg1 = (const float*)d_in[8],  *de1 = (const float*)d_in[9];
    const float* dw2 = (const float*)d_in[10], *db2 = (const float*)d_in[11];
    const float* dg2 = (const float*)d_in[12], *de2 = (const float*)d_in[13];
    const float* dw3 = (const float*)d_in[14], *db3 = (const float*)d_in[15];
    const float* dg3 = (const float*)d_in[16], *de3 = (const float*)d_in[17];
    const float* dw4 = (const float*)d_in[18], *db4 = (const float*)d_in[19];
    float* out = (float*)d_out;

    char* wsb = (char*)d_ws;
    float*          bias_ws = (float*)(wsb + WS_BIAS_OFF);
    unsigned short* wqkv_sw = (unsigned short*)(wsb + WS_WQKV_OFF);
    unsigned short* wout_sw = (unsigned short*)(wsb + WS_WOUT_OFF);

    (void)hipFuncSetAttribute((const void*)attn_win_kernel,
                              hipFuncAttributeMaxDynamicSharedMemorySize, SMEM_BYTES);

    dpb_kernel<<<1, 512, 0, stream>>>(dw1, db1, dg1, de1, dw2, db2, dg2, de2,
                                      dw3, db3, dg3, de3, dw4, db4, bias_ws);
    swizzle_kernel<<<1024, 256, 0, stream>>>(w_qkv, w_out, wqkv_sw, wout_sw);
    attn_win_kernel<<<4096, 256, SMEM_BYTES, stream>>>(x, g, b, b_out,
                                                       bias_ws, wqkv_sw, wout_sw, out);
}
